// TransformerBlock_63900523430581
// MI455X (gfx1250) — matrix operand order repaired and checked by reading, not compiled, not run on hardware
//
#include <hip/hip_runtime.h>
#include <cstdint>
#include <cstddef>

typedef __bf16 bf16;
typedef __bf16 bf16x8  __attribute__((ext_vector_type(8)));
typedef __bf16 bf16x16 __attribute__((ext_vector_type(16)));
typedef float  f32x8   __attribute__((ext_vector_type(8)));
typedef unsigned int u32x4 __attribute__((ext_vector_type(4)));
typedef unsigned int u32x8 __attribute__((ext_vector_type(8)));

namespace {
constexpr int CB = 2, CL = 2048, CD = 1024, CH = 16, CHD = 64, CE = 8, CF = 2048;
constexpr int CT = CB * CL;                       // 4096 tokens
constexpr size_t CTD = (size_t)CT * CD;          // 4,194,304
constexpr size_t CTF = (size_t)CT * CF;          // 8,388,608
}

// ---------------------------------------------------------------- helpers
__device__ __forceinline__ f32x8 wmma_bf16(bf16x16 a, bf16x16 b, f32x8 c) {
  return __builtin_amdgcn_wmma_f32_16x16x32_bf16(false, a, false, b, (short)0, c,
                                                 false, false);
}
__device__ __forceinline__ bf16x16 combine8(bf16x8 lo, bf16x8 hi) {
  bf16x16 r;
#pragma unroll
  for (int i = 0; i < 8; ++i) { r[i] = lo[i]; r[i + 8] = hi[i]; }
  return r;
}
__device__ __forceinline__ f32x8 zero8() {
  f32x8 r;
#pragma unroll
  for (int i = 0; i < 8; ++i) r[i] = 0.f;
  return r;
}
// per-lane async copy of 16 bytes global -> LDS (tracked by ASYNCcnt)
__device__ __forceinline__ void async_b128(unsigned lds_addr, const void* gptr) {
  asm volatile("global_load_async_to_lds_b128 %0, %1, off"
               :: "v"(lds_addr), "v"(gptr)
               : "memory");
}

// ---------------------------------------------------------------- elementwise
__global__ __launch_bounds__(256) void f2bf_kernel(const float* __restrict__ in,
                                                   bf16* __restrict__ out, size_t n) {
  size_t i = (size_t)blockIdx.x * 256 + threadIdx.x;
  if (i < n) out[i] = (bf16)in[i];
}

__global__ __launch_bounds__(256) void silu_mul_kernel(const float* __restrict__ a,
                                                       const float* __restrict__ b,
                                                       bf16* __restrict__ u, size_t n) {
  size_t i = (size_t)blockIdx.x * 256 + threadIdx.x;
  if (i < n) {
    float av = a[i];
    float s  = av / (1.f + __expf(-av));
    u[i] = (bf16)(s * b[i]);
  }
}

// ---------------------------------------------------------------- RMSNorm
__global__ __launch_bounds__(256) void rmsnorm_kernel(const float* __restrict__ x,
                                                      const float* __restrict__ g,
                                                      bf16* __restrict__ out) {
  const int t = blockIdx.x;
  const float* xr = x + (size_t)t * CD;
  __shared__ float red[256];
  float ss = 0.f;
  for (int d = threadIdx.x; d < CD; d += 256) { float v = xr[d]; ss += v * v; }
  red[threadIdx.x] = ss;
  __syncthreads();
  for (int off = 128; off > 0; off >>= 1) {
    if (threadIdx.x < off) red[threadIdx.x] += red[threadIdx.x + off];
    __syncthreads();
  }
  const float rs = rsqrtf(red[0] / (float)CD + 1e-6f);
  for (int d = threadIdx.x; d < CD; d += 256)
    out[(size_t)t * CD + d] = (bf16)(xr[d] * rs * g[d]);
}

// ---------------------------------------------------------------- RoPE + split
__global__ __launch_bounds__(256) void rope_split_kernel(const float* __restrict__ qkv,
                                                         bf16* __restrict__ q,
                                                         bf16* __restrict__ k,
                                                         bf16* __restrict__ v) {
  const int idx = blockIdx.x * 256 + threadIdx.x;  // over B*H*L*HD = 2^22
  const int d  = idx & 63;
  const int l  = (idx >> 6) & (CL - 1);
  const int bh = idx >> 17;
  const int h  = bh & (CH - 1);
  const int b  = bh >> 4;
  const float* base = qkv + ((size_t)(b * CL + l)) * 3 * CD;
  const int col = h * CHD;
  v[idx] = (bf16)base[2 * CD + col + d];
  const int  j      = d & 31;
  const bool second = d >= 32;
  const float inv = powf(10000.f, -(float)(2 * j) / 64.f);
  const float ang = (float)l * inv;
  const float c = cosf(ang), s = sinf(ang);
  const float x1q = base[col + 2 * j],        x2q = base[col + 2 * j + 1];
  const float x1k = base[CD + col + 2 * j],   x2k = base[CD + col + 2 * j + 1];
  q[idx] = (bf16)(second ? (x1q * s + x2q * c) : (x1q * c - x2q * s));
  k[idx] = (bf16)(second ? (x1k * s + x2k * c) : (x1k * c - x2k * s));
}

// ---------------------------------------------------------------- WMMA GEMM
// C[M,N] = A[M,K] * B[N,K]^T ; MODE 0: C=acc, 1: C=acc+R, 2: C += scale[row]*acc
// Block tile 128x128xK32, double-buffered LDS.
//   A tile (128x32): one TDM tensor_load_to_lds issued by wave 0 (TENSORcnt)
//   B tile (128x32): per-lane global_load_async_to_lds_b128 (ASYNCcnt)
template <int MODE>
__global__ __launch_bounds__(256) void gemm_bf16_wmma(const bf16* __restrict__ A,
                                                      const bf16* __restrict__ Bm,
                                                      float* __restrict__ C, int M,
                                                      int N, int K,
                                                      const float* __restrict__ R,
                                                      const float* __restrict__ scale,
                                                      int sstride) {
  __shared__ __align__(16) bf16 As[2][128 * 32];
  __shared__ __align__(16) bf16 Bs[2][128 * 32];
  const int tid  = threadIdx.x;
  const int lane = tid & 31, wave = tid >> 5;
  const int wm = wave >> 1, wn = wave & 1;   // 4 x 2 wave grid
  const int hl = lane & 15, hf = lane >> 4;
  const size_t m0 = (size_t)blockIdx.y * 128;
  const size_t n0 = (size_t)blockIdx.x * 128;

  f32x8 acc[2][4];
#pragma unroll
  for (int i = 0; i < 2; ++i)
#pragma unroll
    for (int j = 0; j < 4; ++j) acc[i][j] = zero8();

  // ---- issue helpers --------------------------------------------------
  auto issue_a_tdm = [&](int buf, int k0) {
    if (wave == 0) {
      const unsigned long long ga =
          (unsigned long long)(uintptr_t)(A + (m0 * (size_t)K + (size_t)k0));
      u32x4 g0;
      g0[0] = 1u;                                         // count=1 valid D#
      g0[1] = (unsigned)(uintptr_t)&As[buf][0];           // lds_addr
      g0[2] = (unsigned)(ga & 0xffffffffu);               // global_addr lo
      g0[3] = (unsigned)((ga >> 32) & 0x1ffffffu) | (2u << 30);  // hi | type=2
      u32x8 g1;
      g1[0] = 1u << 16;                                   // data_size = 2B
      g1[1] = ((unsigned)K & 0xffffu) << 16;              // tensor_dim0 lo
      g1[2] = (((unsigned)K >> 16) & 0xffffu)             // tensor_dim0 hi
            | (((unsigned)M & 0xffffu) << 16);            // tensor_dim1 lo
      g1[3] = (((unsigned)M >> 16) & 0xffffu)             // tensor_dim1 hi
            | (32u << 16);                                // tile_dim0 = 32
      g1[4] = 128u;                                       // tile_dim1=128, dim2=0
      g1[5] = (unsigned)K;                                // dim0_stride lo
      g1[6] = 0u;                                         // dim0_stride hi
      g1[7] = 0u;
      asm volatile("tensor_load_to_lds %0, %1" :: "s"(g0), "s"(g1) : "memory");
    }
  };
  auto issue_b_async = [&](int buf, int k0) {
#pragma unroll
    for (int j = 0; j < 2; ++j) {
      const int chunk = j * 256 + tid;       // 512 chunks of 8 bf16 (16B)
      const int row = chunk >> 2, seg = chunk & 3;
      const bf16* g = &Bm[(n0 + row) * (size_t)K + k0 + seg * 8];
      const unsigned l = (unsigned)(uintptr_t)&Bs[buf][chunk * 8];
      async_b128(l, g);
    }
  };

  const int niter = K >> 5;
  issue_a_tdm(0, 0);
  issue_b_async(0, 0);

  for (int i = 0; i < niter; ++i) {
    const int cur = i & 1, nxt = cur ^ 1;
    const bool more = (i + 1 < niter);
    if (more) {                      // buffers from 2 iters ago are free (barrier)
      issue_a_tdm(nxt, (i + 1) * 32);
      issue_b_async(nxt, (i + 1) * 32);
      asm volatile("s_wait_asynccnt 0x2" ::: "memory");   // cur B done (in-order)
      if (wave == 0) __builtin_amdgcn_s_wait_tensorcnt(1);
    } else {
      asm volatile("s_wait_asynccnt 0x0" ::: "memory");
      if (wave == 0) __builtin_amdgcn_s_wait_tensorcnt(0);
    }
    __syncthreads();   // cur tiles visible to all waves

    bf16x16 af[2], bfr[4];
#pragma unroll
    for (int mt = 0; mt < 2; ++mt) {
      const int r = wm * 32 + mt * 16 + hl;
      bf16x8 lo = *(const bf16x8*)&As[cur][r * 32 + hf * 16];
      bf16x8 hi = *(const bf16x8*)&As[cur][r * 32 + hf * 16 + 8];
      af[mt] = combine8(lo, hi);
    }
#pragma unroll
    for (int nt = 0; nt < 4; ++nt) {
      const int r = wn * 64 + nt * 16 + hl;
      bfr[nt] = *(const bf16x16*)&Bs[cur][r * 32 + hf * 16];
    }
#pragma unroll
    for (int mt = 0; mt < 2; ++mt)
#pragma unroll
      for (int nt = 0; nt < 4; ++nt)
        acc[mt][nt] = wmma_bf16(af[mt], bfr[nt], acc[mt][nt]);
    __syncthreads();   // done reading cur: it may be refilled next iteration
  }

#pragma unroll
  for (int mt = 0; mt < 2; ++mt)
#pragma unroll
    for (int nt = 0; nt < 4; ++nt)
#pragma unroll
      for (int j = 0; j < 8; ++j) {
        const size_t row = m0 + wm * 32 + mt * 16 + j + hf * 8;
        const size_t col = n0 + wn * 64 + nt * 16 + hl;
        const size_t idx = row * (size_t)N + col;
        const float v = acc[mt][nt][j];
        if (MODE == 0) C[idx] = v;
        if (MODE == 1) C[idx] = v + R[idx];
        if (MODE == 2) C[idx] += scale[row * (size_t)sstride] * v;
      }
}

// ---------------------------------------------------------------- attention
// grid: (L/64, B*H), block 128 (4 waves, 16 query rows each). Causal flash attn.
__global__ __launch_bounds__(128) void attn_kernel(const bf16* __restrict__ q,
                                                   const bf16* __restrict__ k,
                                                   const bf16* __restrict__ v,
                                                   bf16* __restrict__ attnO) {
  const int lane = threadIdx.x & 31;
  const int wave = threadIdx.x >> 5;
  const int bh = blockIdx.y;
  const int b = bh >> 4, h = bh & (CH - 1);
  const int q0 = blockIdx.x * 64 + wave * 16;
  const int hl = lane & 15, hf = lane >> 4;

  const bf16* qb = q + (size_t)bh * CL * CHD;
  const bf16* kb = k + (size_t)bh * CL * CHD;
  const bf16* vb = v + (size_t)bh * CL * CHD;

  __shared__ __align__(16) bf16 Vs[4][32 * CHD];
  __shared__ __align__(16) bf16 Ps[4][16 * 32];
  bf16* myV = Vs[wave];
  bf16* myP = Ps[wave];

  // Q fragments (A operand), 2 d-chunks of 32
  bf16x16 qf[2];
#pragma unroll
  for (int dc = 0; dc < 2; ++dc) {
    const int qr = q0 + hl;
    bf16x8 lo = *(const bf16x8*)&qb[(size_t)qr * CHD + dc * 32 + hf * 16];
    bf16x8 hi = *(const bf16x8*)&qb[(size_t)qr * CHD + dc * 32 + hf * 16 + 8];
    qf[dc] = combine8(lo, hi);
  }

  f32x8 o[4];
#pragma unroll
  for (int dt = 0; dt < 4; ++dt) o[dt] = zero8();
  float m_i[8], l_i[8];
#pragma unroll
  for (int j = 0; j < 8; ++j) { m_i[j] = -1e30f; l_i[j] = 0.f; }

  const int kmax = q0 + 16;  // exclusive causal bound for this wave's rows
  for (int kc = 0; kc < kmax; kc += 32) {
    // stage V chunk (32 keys x 64 d) via async global->LDS, one key-row per lane
    {
      const bf16* src = &vb[(size_t)(kc + lane) * CHD];
      const unsigned ldst = (unsigned)(uintptr_t)&myV[lane * CHD];
#pragma unroll
      for (int i = 0; i < 8; ++i) async_b128(ldst + i * 16, src + i * 8);
    }
    // scores: two 16x16 tiles (keys kc..kc+15, kc+16..kc+31)
    f32x8 s0 = zero8(), s1 = zero8();
#pragma unroll
    for (int dc = 0; dc < 2; ++dc) {
      bf16x16 b0 = *(const bf16x16*)&kb[(size_t)(kc + hl) * CHD + dc * 32 + hf * 16];
      bf16x16 b1 = *(const bf16x16*)&kb[(size_t)(kc + 16 + hl) * CHD + dc * 32 + hf * 16];
      s0 = wmma_bf16(qf[dc], b0, s0);
      s1 = wmma_bf16(qf[dc], b1, s1);
    }
    // online softmax per row (row = q0 + j + 8*hf), half-wave reductions
#pragma unroll
    for (int j = 0; j < 8; ++j) {
      const int row = q0 + j + hf * 8;
      float a0 = s0[j] * 0.125f;
      float a1 = s1[j] * 0.125f;
      if (kc + hl > row) a0 = -1e30f;
      if (kc + 16 + hl > row) a1 = -1e30f;
      float mx = fmaxf(a0, a1);
#pragma unroll
      for (int off = 1; off < 16; off <<= 1) mx = fmaxf(mx, __shfl_xor(mx, off, 32));
      const float mnew = fmaxf(m_i[j], mx);
      const float alpha = __expf(m_i[j] - mnew);
      const float e0 = __expf(a0 - mnew);
      const float e1 = __expf(a1 - mnew);
      float rs = e0 + e1;
#pragma unroll
      for (int off = 1; off < 16; off <<= 1) rs += __shfl_xor(rs, off, 32);
      l_i[j] = l_i[j] * alpha + rs;
      m_i[j] = mnew;
#pragma unroll
      for (int dt = 0; dt < 4; ++dt) o[dt][j] = o[dt][j] * alpha;
      myP[(j + hf * 8) * 32 + hl] = (bf16)e0;
      myP[(j + hf * 8) * 32 + 16 + hl] = (bf16)e1;
    }
    // same-wave RAW: async V writes (ASYNCcnt) + P ds stores (DScnt)
    asm volatile("s_wait_asynccnt 0x0\n\ts_wait_dscnt 0x0" ::: "memory");
    // P as A-fragment (16 x 32)
    bf16x8 plo = *(const bf16x8*)&myP[hl * 32 + hf * 16];
    bf16x8 phi = *(const bf16x8*)&myP[hl * 32 + hf * 16 + 8];
    bf16x16 pf = combine8(plo, phi);
    // PV: 4 d-tiles of 16
#pragma unroll
    for (int dt = 0; dt < 4; ++dt) {
      bf16x16 vf;
#pragma unroll
      for (int i = 0; i < 16; ++i) vf[i] = myV[(hf * 16 + i) * CHD + dt * 16 + hl];
      o[dt] = wmma_bf16(pf, vf, o[dt]);
    }
  }
  // normalize + write out as [t, D] with col = h*64 + d
#pragma unroll
  for (int dt = 0; dt < 4; ++dt)
#pragma unroll
    for (int j = 0; j < 8; ++j) {
      const int row = q0 + j + hf * 8;
      const size_t t = (size_t)b * CL + row;
      attnO[t * CD + h * CHD + dt * 16 + hl] = (bf16)(o[dt][j] / l_i[j]);
    }
}

// ---------------------------------------------------------------- MoE gate
__global__ __launch_bounds__(256) void gate_kernel(const bf16* __restrict__ h2,
                                                   const float* __restrict__ wg,
                                                   float* __restrict__ comb,
                                                   float* __restrict__ usage) {
  const int t = blockIdx.x;
  const int tid = threadIdx.x;
  __shared__ float sred[CE][256];
  float accv[CE];
#pragma unroll
  for (int e = 0; e < CE; ++e) accv[e] = 0.f;
  for (int d = tid; d < CD; d += 256) {
    const float hv = (float)h2[(size_t)t * CD + d];
#pragma unroll
    for (int e = 0; e < CE; ++e) accv[e] += hv * wg[(size_t)e * CD + d];
  }
#pragma unroll
  for (int e = 0; e < CE; ++e) sred[e][tid] = accv[e];
  __syncthreads();
  for (int off = 128; off > 0; off >>= 1) {
    if (tid < off) {
#pragma unroll
      for (int e = 0; e < CE; ++e) sred[e][tid] += sred[e][tid + off];
    }
    __syncthreads();
  }
  if (tid == 0) {
    float g[CE];
#pragma unroll
    for (int e = 0; e < CE; ++e) g[e] = sred[e][0];
    float mx = g[0];
#pragma unroll
    for (int e = 1; e < CE; ++e) mx = fmaxf(mx, g[e]);
    float p[CE], sum = 0.f;
#pragma unroll
    for (int e = 0; e < CE; ++e) { p[e] = __expf(g[e] - mx); sum += p[e]; }
#pragma unroll
    for (int e = 0; e < CE; ++e) atomicAdd(&usage[e], p[e] / sum);
    int i0 = 0;
#pragma unroll
    for (int e = 1; e < CE; ++e) if (g[e] > g[i0]) i0 = e;
    int i1 = (i0 == 0) ? 1 : 0;
#pragma unroll
    for (int e = 0; e < CE; ++e) if (e != i0 && g[e] > g[i1]) i1 = e;
    const float w0 = 1.f / (1.f + __expf(g[i1] - g[i0]));
#pragma unroll
    for (int e = 0; e < CE; ++e) comb[(size_t)t * CE + e] = 0.f;
    comb[(size_t)t * CE + i0] = w0;
    comb[(size_t)t * CE + i1] = 1.f - w0;
  }
}

__global__ void aux_kernel(const float* __restrict__ usage, float* __restrict__ dst) {
  if (threadIdx.x == 0 && blockIdx.x == 0) {
    float s = 0.f;
    for (int e = 0; e < CE; ++e) {
      const float u = usage[e] / (float)CT;
      s += u * u;
    }
    *dst = (float)CE * s;
  }
}

// ---------------------------------------------------------------- launcher
extern "C" void kernel_launch(void* const* d_in, const int* in_sizes, int n_in,
                              void* d_out, int out_size, void* d_ws, size_t ws_size,
                              hipStream_t stream) {
  (void)in_sizes; (void)n_in; (void)out_size; (void)ws_size;
  const float* x     = (const float*)d_in[0];
  const float* wqkv  = (const float*)d_in[2];
  const float* wproj = (const float*)d_in[3];
  const float* wgate = (const float*)d_in[4];
  const float* w1    = (const float*)d_in[5];
  const float* w2    = (const float*)d_in[6];
  const float* w3    = (const float*)d_in[7];
  const float* g1    = (const float*)d_in[8];
  const float* g2    = (const float*)d_in[9];
  float* out = (float*)d_out;

  char* p = (char*)d_ws;
  auto carve = [&](size_t bytes) -> void* {
    void* r = (void*)p;
    p += (bytes + 255) & ~(size_t)255;
    return r;
  };
  const size_t EFD = (size_t)CE * CF * CD;
  bf16* h1     = (bf16*)carve(CTD * 2);
  bf16* h2     = (bf16*)carve(CTD * 2);
  bf16* wqkvb  = (bf16*)carve((size_t)3 * CD * CD * 2);
  bf16* wprojb = (bf16*)carve((size_t)CD * CD * 2);
  bf16* w1b    = (bf16*)carve(EFD * 2);
  bf16* w2b    = (bf16*)carve(EFD * 2);
  bf16* w3b    = (bf16*)carve(EFD * 2);
  float* qkv   = (float*)carve((size_t)CT * 3 * CD * 4);
  bf16* qr     = (bf16*)carve(CTD * 2);
  bf16* kr     = (bf16*)carve(CTD * 2);
  bf16* vr     = (bf16*)carve(CTD * 2);
  bf16* attnO  = (bf16*)carve(CTD * 2);
  float* x1    = (float*)carve(CTD * 4);
  float* abuf  = (float*)carve(CTF * 4);
  float* bbuf  = (float*)carve(CTF * 4);
  bf16* ubuf   = (bf16*)carve(CTF * 2);
  float* comb  = (float*)carve((size_t)CT * CE * 4);
  float* usage = (float*)carve(256);

  // 1) rmsnorm(x, g1) -> h1 (bf16)
  rmsnorm_kernel<<<CT, 256, 0, stream>>>(x, g1, h1);
  // 2) weight f32 -> bf16
  {
    const size_t n_qkv = (size_t)3 * CD * CD, n_pj = (size_t)CD * CD;
    f2bf_kernel<<<(int)((n_qkv + 255) / 256), 256, 0, stream>>>(wqkv, wqkvb, n_qkv);
    f2bf_kernel<<<(int)((n_pj + 255) / 256), 256, 0, stream>>>(wproj, wprojb, n_pj);
    f2bf_kernel<<<(int)((EFD + 255) / 256), 256, 0, stream>>>(w1, w1b, EFD);
    f2bf_kernel<<<(int)((EFD + 255) / 256), 256, 0, stream>>>(w2, w2b, EFD);
    f2bf_kernel<<<(int)((EFD + 255) / 256), 256, 0, stream>>>(w3, w3b, EFD);
  }
  // 3) qkv = h1 @ w_qkv^T   (M=4096, N=3072, K=1024)
  gemm_bf16_wmma<0><<<dim3(3 * CD / 128, CT / 128), 256, 0, stream>>>(
      h1, wqkvb, qkv, CT, 3 * CD, CD, nullptr, nullptr, 0);
  // 4) rope + split to [b,h,l,hd] bf16
  rope_split_kernel<<<(CB * CH * CL * CHD) / 256, 256, 0, stream>>>(qkv, qr, kr, vr);
  // 5) causal flash attention -> attnO [t, D] bf16
  attn_kernel<<<dim3(CL / 64, CB * CH), 128, 0, stream>>>(qr, kr, vr, attnO);
  // 6) x1 = x + attnO @ w_proj^T
  gemm_bf16_wmma<1><<<dim3(CD / 128, CT / 128), 256, 0, stream>>>(
      attnO, wprojb, x1, CT, CD, CD, x, nullptr, 0);
  // 7) rmsnorm(x1, g2) -> h2
  rmsnorm_kernel<<<CT, 256, 0, stream>>>(x1, g2, h2);
  // 8) gate: comb + usage
  hipMemsetAsync(usage, 0, CE * sizeof(float), stream);
  gate_kernel<<<CT, 256, 0, stream>>>(h2, wgate, comb, usage);
  aux_kernel<<<1, 32, 0, stream>>>(usage, out + CTD);
  // 9) out starts as x1; MoE experts accumulate into it
  hipMemcpyAsync(out, x1, CTD * sizeof(float), hipMemcpyDeviceToDevice, stream);
  for (int e = 0; e < CE; ++e) {
    const bf16* w1e = w1b + (size_t)e * CF * CD;
    const bf16* w3e = w3b + (size_t)e * CF * CD;
    const bf16* w2e = w2b + (size_t)e * CD * CF;
    gemm_bf16_wmma<0><<<dim3(CF / 128, CT / 128), 256, 0, stream>>>(
        h2, w1e, abuf, CT, CF, CD, nullptr, nullptr, 0);
    gemm_bf16_wmma<0><<<dim3(CF / 128, CT / 128), 256, 0, stream>>>(
        h2, w3e, bbuf, CT, CF, CD, nullptr, nullptr, 0);
    silu_mul_kernel<<<(int)(CTF / 256), 256, 0, stream>>>(abuf, bbuf, ubuf, CTF);
    gemm_bf16_wmma<2><<<dim3(CD / 128, CT / 128), 256, 0, stream>>>(
        ubuf, w2e, out, CT, CD, CF, nullptr, comb + e, CE);
  }
}
